// FeatureAlign_V2_23295902614231
// MI455X (gfx1250) — matrix operand order repaired and checked by reading, not compiled, not run on hardware
//
#include <hip/hip_runtime.h>
#include <hip/hip_bf16.h>
#include <math.h>

typedef __attribute__((ext_vector_type(16))) __bf16 bf16x16;
typedef __attribute__((ext_vector_type(8)))  float  f32x8;
typedef __attribute__((ext_vector_type(4)))  int    i32x4;

typedef __attribute__((address_space(1))) i32x4* as1_i32x4_p;
typedef __attribute__((address_space(3))) i32x4* as3_i32x4_p;

#define WMMA_BF16(a, b, c) \
  __builtin_amdgcn_wmma_f32_16x16x32_bf16(false, (a), false, (b), (short)0, (c), false, false)

#define Cc   128
#define Hh   128
#define Ww   128
#define Pp   (Hh * Ww)     // 16384
#define Bb   2
#define OMC  216           // dg*3*kk
#define KDCN 1152          // 128 * 9
#define OMW_PADO 224       // OMC padded to multiple of 16

#if defined(__has_builtin)
#if __has_builtin(__builtin_amdgcn_global_load_async_to_lds_b128)
#define HAS_ASYNC_LDS 1
#endif
#endif

__device__ __forceinline__ float sigmoidf_(float x) { return 1.0f / (1.0f + __expf(-x)); }

// ---------------- SE path ----------------
__global__ __launch_bounds__(256) void pool_kernel(const float* __restrict__ feat_l,
                                                   float* __restrict__ pooled) {
  __shared__ float red[256];
  int bc = blockIdx.x;                       // 0..255 = b*C + c
  const float* src = feat_l + (size_t)bc * Pp;
  float s = 0.f;
  for (int i = threadIdx.x; i < Pp; i += 256) s += src[i];
  red[threadIdx.x] = s;
  __syncthreads();
  for (int o = 128; o > 0; o >>= 1) {
    if (threadIdx.x < o) red[threadIdx.x] += red[threadIdx.x + o];
    __syncthreads();
  }
  if (threadIdx.x == 0) pooled[bc] = red[0] * (1.0f / (float)Pp);
}

__global__ __launch_bounds__(256) void atten_kernel(const float* __restrict__ pooled,
    const float* __restrict__ atten_w, const float* __restrict__ atten_b,
    const float* __restrict__ bn_g, const float* __restrict__ bn_b,
    const float* __restrict__ bn_m, const float* __restrict__ bn_v,
    float* __restrict__ atten) {
  int t = threadIdx.x;                       // single block of 256
  int b = t >> 7, o = t & 127;
  float a = atten_b[o];
  for (int c = 0; c < Cc; ++c) a += pooled[b * Cc + c] * atten_w[o * Cc + c];
  a = (a - bn_m[o]) * rsqrtf(bn_v[o] + 1e-5f) * bn_g[o] + bn_b[o];
  atten[t] = sigmoidf_(a);
}

// ---------------- bilinear 2x upsample (align_corners) ----------------
__global__ __launch_bounds__(128) void upsample_kernel(const float* __restrict__ feat_s,
                                                       float* __restrict__ feat_up) {
  int id = blockIdx.x;                       // bc*128 + h
  int h = id & 127;
  int bc = id >> 7;
  int w = threadIdx.x;
  const float sc = 63.0f / 127.0f;
  float cy = h * sc, cx = w * sc;
  int y0 = (int)floorf(cy); int y1 = min(y0 + 1, 63); float wy = cy - (float)y0;
  int x0 = (int)floorf(cx); int x1 = min(x0 + 1, 63); float wx = cx - (float)x0;
  const float* src = feat_s + (size_t)bc * 64 * 64;
  float v = src[y0 * 64 + x0] * (1.f - wy) * (1.f - wx)
          + src[y0 * 64 + x1] * (1.f - wy) * wx
          + src[y1 * 64 + x0] * wy * (1.f - wx)
          + src[y1 * 64 + x1] * wy * wx;
  feat_up[(size_t)bc * Pp + h * Ww + w] = v;
}

// ---------------- weight prep (bf16; transpose for 3x3 head) ----------------
__global__ __launch_bounds__(256) void prep_omw_kernel(const float* __restrict__ om_w,
                                                       __bf16* __restrict__ om_wt) {
  int idx = blockIdx.x * 256 + threadIdx.x;
  if (idx >= 9 * OMW_PADO * Cc) return;
  int kk = idx / (OMW_PADO * Cc);
  int rem = idx % (OMW_PADO * Cc);
  int o = rem / Cc, c = rem % Cc;
  float v = (o < OMC) ? om_w[((size_t)(o * Cc + c)) * 9 + kk] : 0.0f;
  om_wt[idx] = (__bf16)v;
}

__global__ __launch_bounds__(256) void prep_dcnw_kernel(const float* __restrict__ dcn_w,
                                                        __bf16* __restrict__ dcn_wb) {
  int idx = blockIdx.x * 256 + threadIdx.x;
  if (idx >= Cc * KDCN) return;
  dcn_wb[idx] = (__bf16)dcn_w[idx];
}

__global__ __launch_bounds__(256) void prep_small_kernel(const float* __restrict__ fsm_w,
    const float* __restrict__ off_w, __bf16* __restrict__ fsm_wb, __bf16* __restrict__ off_wb) {
  int idx = blockIdx.x * 256 + threadIdx.x;
  if (idx < Cc * Cc) {
    fsm_wb[idx] = (__bf16)fsm_w[idx];
  } else if (idx < Cc * Cc + Cc * 256) {
    int j = idx - Cc * Cc;
    off_wb[j] = (__bf16)off_w[j];
  }
}

// ---------------- fsm: 1x1 conv as WMMA GEMM (M=128,K=128,N=16 per WG) ----------------
__global__ __launch_bounds__(256) void fsm_gemm_kernel(const float* __restrict__ feat_l,
    const float* __restrict__ atten, const __bf16* __restrict__ fsm_wb,
    const float* __restrict__ fsm_b, float* __restrict__ feat_arm) {
  __shared__ __bf16 ldsB[16 * Cc];           // [n][k]
  int b = blockIdx.y;
  int p0 = blockIdx.x * 16;
  int t = threadIdx.x;
  {
    int k = t >> 1;
    int n0 = (t & 1) * 8;
    float sc = 1.0f + atten[b * Cc + k];     // x + x*atten == x*(1+atten)
    const float* src = feat_l + (size_t)(b * Cc + k) * Pp + p0 + n0;
#pragma unroll
    for (int i = 0; i < 8; ++i) ldsB[(n0 + i) * Cc + k] = (__bf16)(src[i] * sc);
  }
  __syncthreads();
  int lane = t & 31, wave = t >> 5;
  int half = lane >> 4, r = lane & 15;
  int m0 = wave * 16;
  f32x8 acc = {};
#pragma unroll
  for (int kt = 0; kt < 4; ++kt) {
    int k0 = kt * 32;
    bf16x16 afrag, bfrag;
    const __bf16* wrow = fsm_wb + (size_t)(m0 + r) * Cc + k0;
#pragma unroll
    for (int e = 0; e < 8; ++e) {
      afrag[e]     = wrow[half * 16 + e];
      afrag[e + 8] = wrow[half * 16 + 8 + e];
    }
    const __bf16* bcol = &ldsB[r * Cc + k0 + half * 16];
#pragma unroll
    for (int e = 0; e < 16; ++e) bfrag[e] = bcol[e];
    acc = WMMA_BF16(afrag, bfrag, acc);
  }
#pragma unroll
  for (int rr = 0; rr < 8; ++rr) {
    int m = m0 + rr + half * 8;
    feat_arm[(size_t)(b * Cc + m) * Pp + p0 + r] = acc[rr] + fsm_b[m];
  }
}

// ---------------- offset feature: 1x1 conv, K=256, bf16 channel-last output ----------------
__global__ __launch_bounds__(256) void off_gemm_kernel(const float* __restrict__ feat_arm,
    const float* __restrict__ feat_up, const __bf16* __restrict__ off_wb,
    __bf16* __restrict__ off_feat /* [b][p][c] */) {
  __shared__ __bf16 ldsB[16 * 256];          // [n][k]
  int b = blockIdx.y;
  int p0 = blockIdx.x * 16;
  int t = threadIdx.x;
  {
    int k = t;
    const float* src = (k < Cc) ? (feat_arm + (size_t)(b * Cc + k) * Pp + p0)
                                : (feat_up + (size_t)(b * Cc + (k - Cc)) * Pp + p0);
#pragma unroll
    for (int n2 = 0; n2 < 16; ++n2) ldsB[n2 * 256 + k] = (__bf16)src[n2];
  }
  __syncthreads();
  int lane = t & 31, wave = t >> 5;
  int half = lane >> 4, r = lane & 15;
  int m0 = wave * 16;
  f32x8 acc = {};
#pragma unroll
  for (int kt = 0; kt < 8; ++kt) {
    int k0 = kt * 32;
    bf16x16 afrag, bfrag;
    const __bf16* wrow = off_wb + (size_t)(m0 + r) * 256 + k0;
#pragma unroll
    for (int e = 0; e < 8; ++e) {
      afrag[e]     = wrow[half * 16 + e];
      afrag[e + 8] = wrow[half * 16 + 8 + e];
    }
    const __bf16* bcol = &ldsB[r * 256 + k0 + half * 16];
#pragma unroll
    for (int e = 0; e < 16; ++e) bfrag[e] = bcol[e];
    acc = WMMA_BF16(afrag, bfrag, acc);
  }
#pragma unroll
  for (int rr = 0; rr < 8; ++rr) {
    int m = m0 + rr + half * 8;
    off_feat[((size_t)b * Pp + p0 + r) * Cc + m] = (__bf16)acc[rr];
  }
}

// ---------------- offset/mask head: 3x3 conv as implicit WMMA GEMM ----------------
__global__ __launch_bounds__(256) void om_conv_kernel(const __bf16* __restrict__ off_feat,
    const __bf16* __restrict__ om_wt /* [kk][224][128] */, const float* __restrict__ om_b,
    float* __restrict__ om /* [b][216][p], mask channels pre-sigmoided */) {
  __shared__ __attribute__((aligned(16))) __bf16 ldsX[3 * 18 * Cc]; // [(row*18+col)][c]
  int b = blockIdx.y;
  int tile = blockIdx.x;
  int h = tile >> 3;
  int w0 = (tile & 7) << 4;
  int t = threadIdx.x;
#ifdef HAS_ASYNC_LDS
  {
    // 54 halo cells x 256B each, copied as 16B chunks: global bf16 -> LDS, async.
    for (int idx = t; idx < 54 * 16; idx += 256) {
      int cell = idx >> 4;
      int bo = (idx & 15) * 16;              // byte offset within cell
      int row = cell / 18, col = cell % 18;
      int gh = h - 1 + row, gw = w0 - 1 + col;
      char* ldst = ((char*)ldsX) + cell * 256 + bo;
      if (gh >= 0 && gh < Hh && gw >= 0 && gw < Ww) {
        char* gsrc = const_cast<char*>(((const char*)off_feat) +
                      (((size_t)b * Pp + gh * Ww + gw) * Cc) * sizeof(__bf16) + bo);
        __builtin_amdgcn_global_load_async_to_lds_b128(
            (as1_i32x4_p)gsrc, (as3_i32x4_p)ldst, 0, 0);
      } else {
        *(i32x4*)ldst = (i32x4){0, 0, 0, 0};
      }
    }
#if defined(__has_builtin) && __has_builtin(__builtin_amdgcn_s_wait_asynccnt)
    __builtin_amdgcn_s_wait_asynccnt(0);
#else
    asm volatile("s_wait_asynccnt 0x0" ::: "memory");
#endif
  }
#else
  if (t < 216) {
    int cell = t >> 2;                       // 0..53 (3 rows x 18 cols)
    int chunk = t & 3;                       // 32 channels each
    int row = cell / 18, col = cell % 18;
    int gh = h - 1 + row, gw = w0 - 1 + col;
    __bf16* dst = &ldsX[cell * Cc + chunk * 32];
    if (gh >= 0 && gh < Hh && gw >= 0 && gw < Ww) {
      const __bf16* src = off_feat + ((size_t)b * Pp + gh * Ww + gw) * Cc + chunk * 32;
#pragma unroll
      for (int i = 0; i < 32; ++i) dst[i] = src[i];
    } else {
#pragma unroll
      for (int i = 0; i < 32; ++i) dst[i] = (__bf16)0.0f;
    }
  }
#endif
  __syncthreads();
  int lane = t & 31, wave = t >> 5;
  int half = lane >> 4, r = lane & 15;
  int m0a = wave * 16;
  int m0b = (wave + 8) * 16;
  bool doB = (m0b < OMC);                    // wave-uniform
  f32x8 acc0 = {}, acc1 = {};
  for (int kk = 0; kk < 9; ++kk) {
    int ky = kk / 3, kx = kk % 3;
#pragma unroll
    for (int kt = 0; kt < 4; ++kt) {
      int k0 = kt * 32;
      bf16x16 bfrag;
      const __bf16* bcol = &ldsX[(ky * 18 + r + kx) * Cc + k0 + half * 16];
#pragma unroll
      for (int e = 0; e < 16; ++e) bfrag[e] = bcol[e];
      bf16x16 a0;
      const __bf16* wp0 = om_wt + ((size_t)kk * OMW_PADO + (m0a + r)) * Cc + k0;
#pragma unroll
      for (int e = 0; e < 8; ++e) { a0[e] = wp0[half * 16 + e]; a0[e + 8] = wp0[half * 16 + 8 + e]; }
      acc0 = WMMA_BF16(a0, bfrag, acc0);
      if (doB) {
        bf16x16 a1;
        const __bf16* wp1 = om_wt + ((size_t)kk * OMW_PADO + (m0b + r)) * Cc + k0;
#pragma unroll
        for (int e = 0; e < 8; ++e) { a1[e] = wp1[half * 16 + e]; a1[e + 8] = wp1[half * 16 + 8 + e]; }
        acc1 = WMMA_BF16(a1, bfrag, acc1);
      }
    }
  }
  int p = h * Ww + w0 + r;
#pragma unroll
  for (int rr = 0; rr < 8; ++rr) {
    int m = m0a + rr + half * 8;             // always < 128 -> offset channels
    om[(size_t)(b * OMC + m) * Pp + p] = acc0[rr] + om_b[m];
  }
  if (doB) {
#pragma unroll
    for (int rr = 0; rr < 8; ++rr) {
      int m = m0b + rr + half * 8;
      if (m < OMC) {
        float v = acc1[rr] + om_b[m];
        if (m >= 144) v = sigmoidf_(v);      // mask channels
        om[(size_t)(b * OMC + m) * Pp + p] = v;
      }
    }
  }
}

// ---------------- DCNv2: gather->LDS (bf16) then K=1152 WMMA GEMM + epilogue ----------------
__global__ __launch_bounds__(256) void dcn_kernel(const float* __restrict__ feat_up,
    const float* __restrict__ om, const __bf16* __restrict__ dcn_wb /* [128][1152] */,
    const float* __restrict__ dcn_b, const float* __restrict__ feat_arm,
    float* __restrict__ out) {
  __shared__ __attribute__((aligned(16))) __bf16 ldsS[16 * KDCN]; // [n][(g*16+c)*9+kk]
  int b = blockIdx.y;
  int tile = blockIdx.x;
  int h = tile >> 3;
  int w0 = (tile & 7) << 4;
  int t = threadIdx.x;
  {
    int n = t & 15;
    int j = t >> 4;
    int w = w0 + n;
    int p = h * Ww + w;
    for (int gk = j; gk < 72; gk += 16) {
      int g = gk / 9, kk = gk % 9;
      float oy = om[((size_t)(b * OMC) + gk) * Pp + p];
      float ox = om[((size_t)(b * OMC) + 72 + gk) * Pp + p];
      float mk = om[((size_t)(b * OMC) + 144 + gk) * Pp + p];
      float py = (float)h + (float)(kk / 3 - 1) + oy;
      float px = (float)w + (float)(kk % 3 - 1) + ox;
      float y0f = floorf(py), x0f = floorf(px);
      float ly = py - y0f, lx = px - x0f;
      int y0 = (int)y0f, x0 = (int)x0f;
      int y1 = y0 + 1, x1 = x0 + 1;
      float w00 = (1.f - ly) * (1.f - lx), w01 = (1.f - ly) * lx;
      float w10 = ly * (1.f - lx),         w11 = ly * lx;
      bool vy0 = (y0 >= 0) & (y0 < Hh), vy1 = (y1 >= 0) & (y1 < Hh);
      bool vx0 = (x0 >= 0) & (x0 < Ww), vx1 = (x1 >= 0) & (x1 < Ww);
      if (!(vy0 && vx0)) w00 = 0.f;
      if (!(vy0 && vx1)) w01 = 0.f;
      if (!(vy1 && vx0)) w10 = 0.f;
      if (!(vy1 && vx1)) w11 = 0.f;
      int cy0 = min(max(y0, 0), Hh - 1), cy1 = min(max(y1, 0), Hh - 1);
      int cx0 = min(max(x0, 0), Ww - 1), cx1 = min(max(x1, 0), Ww - 1);
      int i00 = cy0 * Ww + cx0, i01 = cy0 * Ww + cx1;
      int i10 = cy1 * Ww + cx0, i11 = cy1 * Ww + cx1;
      const float* base = feat_up + (size_t)(b * Cc + g * 16) * Pp;
      __bf16* dst = &ldsS[n * KDCN + (g * 16) * 9 + kk];
#pragma unroll
      for (int c = 0; c < 16; ++c) {
        const float* ch = base + (size_t)c * Pp;
        float v = w00 * ch[i00] + w01 * ch[i01] + w10 * ch[i10] + w11 * ch[i11];
        dst[c * 9] = (__bf16)(v * mk);
      }
    }
  }
  __syncthreads();
  int lane = t & 31, wave = t >> 5;
  int half = lane >> 4, r = lane & 15;
  int m0 = wave * 16;
  f32x8 acc = {};
  for (int kt = 0; kt < 36; ++kt) {
    int k0 = kt * 32;
    bf16x16 afrag, bfrag;
    const __bf16* wrow = dcn_wb + (size_t)(m0 + r) * KDCN + k0;
    __builtin_prefetch(wrow + 128, 0, 1);   // stream next weight chunk (global_prefetch_b8)
#pragma unroll
    for (int e = 0; e < 8; ++e) { afrag[e] = wrow[half * 16 + e]; afrag[e + 8] = wrow[half * 16 + 8 + e]; }
    const __bf16* bcol = &ldsS[r * KDCN + k0 + half * 16];
#pragma unroll
    for (int e = 0; e < 16; ++e) bfrag[e] = bcol[e];
    acc = WMMA_BF16(afrag, bfrag, acc);
  }
  int p = h * Ww + w0 + r;
#pragma unroll
  for (int rr = 0; rr < 8; ++rr) {
    int m = m0 + rr + half * 8;
    float v = fmaxf(acc[rr] + dcn_b[m], 0.0f);
    out[(size_t)(b * Cc + m) * Pp + p] = v + feat_arm[(size_t)(b * Cc + m) * Pp + p];
  }
}

extern "C" void kernel_launch(void* const* d_in, const int* in_sizes, int n_in,
                              void* d_out, int out_size, void* d_ws, size_t ws_size,
                              hipStream_t stream) {
  const float* feat_l  = (const float*)d_in[0];
  const float* feat_s  = (const float*)d_in[1];
  const float* atten_w = (const float*)d_in[2];
  const float* atten_b = (const float*)d_in[3];
  const float* bn_g    = (const float*)d_in[4];
  const float* bn_b    = (const float*)d_in[5];
  const float* bn_m    = (const float*)d_in[6];
  const float* bn_v    = (const float*)d_in[7];
  const float* fsm_w   = (const float*)d_in[8];
  const float* fsm_b   = (const float*)d_in[9];
  const float* off_w   = (const float*)d_in[10];
  const float* om_w    = (const float*)d_in[11];
  const float* om_b    = (const float*)d_in[12];
  const float* dcn_w   = (const float*)d_in[13];
  const float* dcn_b   = (const float*)d_in[14];
  float* out = (float*)d_out;
  (void)in_sizes; (void)n_in; (void)out_size; (void)ws_size;

  char* ws = (char*)d_ws;
  size_t off = 0;
  auto alloc = [&](size_t bytes) {
    void* p = ws + off;
    off = (off + bytes + 255) & ~(size_t)255;
    return p;
  };
  float*  pooled   = (float*)alloc((size_t)Bb * Cc * sizeof(float));
  float*  atten    = (float*)alloc((size_t)Bb * Cc * sizeof(float));
  float*  feat_up  = (float*)alloc((size_t)Bb * Cc * Pp * sizeof(float));
  float*  feat_arm = (float*)alloc((size_t)Bb * Cc * Pp * sizeof(float));
  float*  om       = (float*)alloc((size_t)Bb * OMC * Pp * sizeof(float));
  __bf16* off_feat = (__bf16*)alloc((size_t)Bb * Pp * Cc * sizeof(__bf16));
  __bf16* om_wt    = (__bf16*)alloc((size_t)9 * OMW_PADO * Cc * sizeof(__bf16));
  __bf16* dcn_wb   = (__bf16*)alloc((size_t)Cc * KDCN * sizeof(__bf16));
  __bf16* fsm_wb   = (__bf16*)alloc((size_t)Cc * Cc * sizeof(__bf16));
  __bf16* off_wb   = (__bf16*)alloc((size_t)Cc * 256 * sizeof(__bf16));

  pool_kernel<<<Bb * Cc, 256, 0, stream>>>(feat_l, pooled);
  atten_kernel<<<1, 256, 0, stream>>>(pooled, atten_w, atten_b, bn_g, bn_b, bn_m, bn_v, atten);
  upsample_kernel<<<Bb * Cc * Hh, 128, 0, stream>>>(feat_s, feat_up);
  prep_omw_kernel<<<(9 * OMW_PADO * Cc + 255) / 256, 256, 0, stream>>>(om_w, om_wt);
  prep_dcnw_kernel<<<(Cc * KDCN + 255) / 256, 256, 0, stream>>>(dcn_w, dcn_wb);
  prep_small_kernel<<<(Cc * Cc + Cc * 256 + 255) / 256, 256, 0, stream>>>(fsm_w, off_w, fsm_wb, off_wb);

  dim3 g(Pp / 16, Bb);
  fsm_gemm_kernel<<<g, 256, 0, stream>>>(feat_l, atten, fsm_wb, fsm_b, feat_arm);
  off_gemm_kernel<<<g, 256, 0, stream>>>(feat_arm, feat_up, off_wb, off_feat);
  om_conv_kernel<<<g, 256, 0, stream>>>(off_feat, om_wt, om_b, om);
  dcn_kernel<<<g, 256, 0, stream>>>(feat_up, om, dcn_wb, dcn_b, feat_arm, out);
}
